// MultiHeadAttention_19086834663607
// MI455X (gfx1250) — matrix operand order repaired and checked by reading, not compiled, not run on hardware
//
#include <hip/hip_runtime.h>
#include <hip/hip_bf16.h>

#define B_  2
#define S_  2048
#define D_  1024
#define H_  16
#define DK_ 64
#define NEGV (-1000000000.0f)

typedef __attribute__((ext_vector_type(16))) __bf16 bf16x16;
typedef __attribute__((ext_vector_type(8)))  __bf16 bf16x8;
typedef __attribute__((ext_vector_type(8)))  float  f32x8;
typedef __attribute__((ext_vector_type(4)))  float  f32x4;

// ---------------------------------------------------------------------------
// NT GEMM: C[m,n] = sum_k A[m*lda+k] * Bm[n*ldb+k]. WMMA operands are bf16,
// accumulate fp32. Overloaded fragment loaders: fp32 source (cvt in-register)
// or bf16 source (straight b128 loads, zero conversion VALU).
//
// A 16x32 bf16 layout (ISA 7.12.2): lane = {m = lane&15, hi = lane>>4}
//   element e<8 : K = 8*hi + e          element e>=8: K = 16 + 8*hi + (e-8)
// B 32x16 bf16 layout: lane = {n = lane&15, hi = lane>>4}, elem e: K as for A (e<8: 8*hi+e, e>=8: 16+8*hi+(e-8))
// C/D f32 16x16: VGPR r -> M = r + 8*hi, N = lane&15
// ---------------------------------------------------------------------------
__device__ __forceinline__ bf16x16 load_a_frag(const float* __restrict__ A,
                                               long lda, long m0, long k0, int lane) {
  const int m = lane & 15, hi = lane >> 4;
  const float* row = A + (m0 + m) * lda + k0;
  f32x4 r0 = *reinterpret_cast<const f32x4*>(row + 8 * hi);
  f32x4 r1 = *reinterpret_cast<const f32x4*>(row + 8 * hi + 4);
  f32x4 r2 = *reinterpret_cast<const f32x4*>(row + 16 + 8 * hi);
  f32x4 r3 = *reinterpret_cast<const f32x4*>(row + 16 + 8 * hi + 4);
  bf16x16 a;
#pragma unroll
  for (int e = 0; e < 4; ++e) {
    a[e] = (__bf16)r0[e]; a[4 + e] = (__bf16)r1[e];
    a[8 + e] = (__bf16)r2[e]; a[12 + e] = (__bf16)r3[e];
  }
  return a;
}

__device__ __forceinline__ bf16x16 load_a_frag(const __bf16* __restrict__ A,
                                               long lda, long m0, long k0, int lane) {
  const int m = lane & 15, hi = lane >> 4;
  const __bf16* row = A + (m0 + m) * lda + k0;
  bf16x8 lo = *reinterpret_cast<const bf16x8*>(row + 8 * hi);
  bf16x8 hv = *reinterpret_cast<const bf16x8*>(row + 16 + 8 * hi);
  bf16x16 a;
#pragma unroll
  for (int e = 0; e < 8; ++e) { a[e] = lo[e]; a[8 + e] = hv[e]; }
  return a;
}

__device__ __forceinline__ bf16x16 load_b_frag(const float* __restrict__ Bm,
                                               long ldb, long n0, long k0, int lane) {
  const int n = lane & 15, hi = lane >> 4;
  const float* row = Bm + (n0 + n) * ldb + k0 + 8 * hi;
  bf16x16 b;
#pragma unroll
  for (int c = 0; c < 4; ++c) {
    f32x4 r = *reinterpret_cast<const f32x4*>(row + 4 * c + 8 * (c >> 1));
#pragma unroll
    for (int e = 0; e < 4; ++e) b[4 * c + e] = (__bf16)r[e];
  }
  return b;
}

__device__ __forceinline__ bf16x16 load_b_frag(const __bf16* __restrict__ Bm,
                                               long ldb, long n0, long k0, int lane) {
  const int n = lane & 15, hi = lane >> 4;
  const __bf16* row = Bm + (n0 + n) * ldb + k0 + 8 * hi;
  bf16x8 b0 = *reinterpret_cast<const bf16x8*>(row);
  bf16x8 b1 = *reinterpret_cast<const bf16x8*>(row + 16);
  bf16x16 b;
#pragma unroll
  for (int e = 0; e < 8; ++e) { b[e] = b0[e]; b[8 + e] = b1[e]; }
  return b;
}

// One wave computes a 32(M) x 64(N) tile: 8 WMMAs per K-step from 6 fragment
// loads (2 A + 4 B); A reused 4x, B reused 2x.
template <typename TA, typename TB>
__device__ __forceinline__ void gemm_tile_32x64(const TA* __restrict__ A, long lda,
                                                const TB* __restrict__ Bm, long ldb,
                                                long m0, long n0, int K, int lane,
                                                f32x8 c[2][4]) {
  for (int k0 = 0; k0 < K; k0 += 32) {
    bf16x16 a0 = load_a_frag(A, lda, m0, (long)k0, lane);
    bf16x16 a1 = load_a_frag(A, lda, m0 + 16, (long)k0, lane);
#pragma unroll
    for (int t = 0; t < 4; ++t) {
      bf16x16 b = load_b_frag(Bm, ldb, n0 + 16 * t, (long)k0, lane);
      c[0][t] = __builtin_amdgcn_wmma_f32_16x16x32_bf16(
          false, a0, false, b, (short)0, c[0][t], false, false);
      c[1][t] = __builtin_amdgcn_wmma_f32_16x16x32_bf16(
          false, a1, false, b, (short)0, c[1][t], false, false);
    }
  }
}

// ---------------------------------------------------------------------------
// Projections: Y = X @ W^T + bias, scaled, stored bf16 in per-head layouts.
// mode 0/1: dst[(b*H+h)*S*DK + s*DK + dk]   (q pre-scaled by 1/sqrt(DK))
// mode 2  : dst[(b*H+h)*DK*S + dk*S + s]    (V transposed for attn@V)
// ---------------------------------------------------------------------------
__global__ void proj_kernel(const float* __restrict__ X, const float* __restrict__ W,
                            const float* __restrict__ bias, __bf16* __restrict__ dst,
                            int mode, float scale) {
  const int lane = threadIdx.x & 31;
  const long n0 = (long)blockIdx.x * 64;
  const long m0 = (long)blockIdx.y * 32;
  f32x8 c[2][4] = {};
  gemm_tile_32x64(X, (long)D_, W, (long)D_, m0, n0, D_, lane, c);
  const int nn = lane & 15, hi = lane >> 4;
#pragma unroll
  for (int t = 0; t < 4; ++t) {
    const long n = n0 + 16 * t + nn;
    const float bv = bias[n];
    const long h = n >> 6, dk = n & (DK_ - 1);
#pragma unroll
    for (int mi = 0; mi < 2; ++mi)
#pragma unroll
      for (int r = 0; r < 8; ++r) {
        const long m = m0 + 16 * mi + r + 8 * hi;
        const long b = m >> 11, s = m & (S_ - 1);
        const long bh = b * H_ + h;
        const __bf16 val = (__bf16)((c[mi][t][r] + bv) * scale);
        if (mode == 2) dst[(bh * DK_ + dk) * S_ + s] = val;
        else           dst[(bh * (long)S_ + s) * DK_ + dk] = val;
      }
  }
}

// scores[bh, i, j] = q[bh,i,:] . k[bh,j,:]  (fp32 out, straight into d_out)
__global__ void scores_kernel(const __bf16* __restrict__ qs, const __bf16* __restrict__ ks,
                              float* __restrict__ attn) {
  const int lane = threadIdx.x & 31;
  const long bh = blockIdx.z;
  const long n0 = (long)blockIdx.x * 64;
  const long m0 = (long)blockIdx.y * 32;
  const __bf16* A  = qs + bh * (long)S_ * DK_;
  const __bf16* Bm = ks + bh * (long)S_ * DK_;
  f32x8 c[2][4] = {};
  gemm_tile_32x64(A, (long)DK_, Bm, (long)DK_, m0, n0, DK_, lane, c);
  float* out = attn + bh * (long)S_ * S_;
  const int nn = lane & 15, hi = lane >> 4;
#pragma unroll
  for (int t = 0; t < 4; ++t)
#pragma unroll
    for (int mi = 0; mi < 2; ++mi)
#pragma unroll
      for (int r = 0; r < 8; ++r)
        out[(m0 + 16 * mi + r + 8 * hi) * (long)S_ + n0 + 16 * t + nn] = c[mi][t][r];
}

// One wave per row: masked-fill(-1e9) + numerically stable softmax, in place.
__global__ void softmax_kernel(float* __restrict__ attn, const unsigned char* __restrict__ mask) {
  const int lane = threadIdx.x & 31;
  const long rowIdx = (long)blockIdx.x * (blockDim.x >> 5) + (threadIdx.x >> 5);
  if (rowIdx >= (long)B_ * H_ * S_) return;
  const long bh = rowIdx / S_;
  const long q  = rowIdx - bh * S_;
  const long b  = bh / H_;
  float* row = attn + rowIdx * (long)S_;
  const unsigned char* mrow = mask + (b * S_ + q) * (long)S_;
  float r[S_ / 32];
  float mx = -__builtin_inff();
#pragma unroll
  for (int i = 0; i < S_ / 32; ++i) {
    const int col = lane + 32 * i;
    float v = row[col];
    if (mrow[col]) v = NEGV;
    r[i] = v;
    mx = fmaxf(mx, v);
  }
#pragma unroll
  for (int off = 16; off > 0; off >>= 1) mx = fmaxf(mx, __shfl_xor(mx, off, 32));
  float sum = 0.0f;
#pragma unroll
  for (int i = 0; i < S_ / 32; ++i) { r[i] = __expf(r[i] - mx); sum += r[i]; }
#pragma unroll
  for (int off = 16; off > 0; off >>= 1) sum += __shfl_xor(sum, off, 32);
  const float inv = 1.0f / sum;
#pragma unroll
  for (int i = 0; i < S_ / 32; ++i) row[lane + 32 * i] = r[i] * inv;
}

// attn_out[b, s, h*DK+dk] = sum_k attn[bh,s,k] * vT[bh,dk,k]  (bf16 out to ws)
__global__ void attnv_kernel(const float* __restrict__ attn, const __bf16* __restrict__ vT,
                             __bf16* __restrict__ ao) {
  const int lane = threadIdx.x & 31;
  const long bh = blockIdx.z;
  const long m0 = (long)blockIdx.y * 32;
  const float*  A  = attn + bh * (long)S_ * S_;
  const __bf16* Bm = vT + bh * (long)S_ * DK_;   // [DK, S] row-major
  f32x8 c[2][4] = {};
  gemm_tile_32x64(A, (long)S_, Bm, (long)S_, m0, 0L, S_, lane, c);
  const long b = bh / H_, h = bh - b * H_;
  const int nn = lane & 15, hi = lane >> 4;
#pragma unroll
  for (int t = 0; t < 4; ++t)
#pragma unroll
    for (int mi = 0; mi < 2; ++mi)
#pragma unroll
      for (int r = 0; r < 8; ++r) {
        const long s = m0 + 16 * mi + r + 8 * hi;
        ao[(b * S_ + s) * (long)D_ + h * DK_ + 16 * t + nn] = (__bf16)c[mi][t][r];
      }
}

// output = attn_out @ fc_w^T + fc_b  (fp32 out)
__global__ void fc_kernel(const __bf16* __restrict__ ao, const float* __restrict__ W,
                          const float* __restrict__ bias, float* __restrict__ out) {
  const int lane = threadIdx.x & 31;
  const long n0 = (long)blockIdx.x * 64;
  const long m0 = (long)blockIdx.y * 32;
  f32x8 c[2][4] = {};
  gemm_tile_32x64(ao, (long)D_, W, (long)D_, m0, n0, D_, lane, c);
  const int nn = lane & 15, hi = lane >> 4;
#pragma unroll
  for (int t = 0; t < 4; ++t) {
    const long n = n0 + 16 * t + nn;
    const float bv = bias[n];
#pragma unroll
    for (int mi = 0; mi < 2; ++mi)
#pragma unroll
      for (int r = 0; r < 8; ++r)
        out[(m0 + 16 * mi + r + 8 * hi) * (long)D_ + n] = c[mi][t][r] + bv;
  }
}

extern "C" void kernel_launch(void* const* d_in, const int* in_sizes, int n_in,
                              void* d_out, int out_size, void* d_ws, size_t ws_size,
                              hipStream_t stream) {
  (void)in_sizes; (void)n_in; (void)out_size; (void)ws_size;
  const float* Q    = (const float*)d_in[0];
  const float* K    = (const float*)d_in[1];
  const float* V    = (const float*)d_in[2];
  const unsigned char* mask = (const unsigned char*)d_in[3];
  const float* WQ_w = (const float*)d_in[4];
  const float* WQ_b = (const float*)d_in[5];
  const float* WK_w = (const float*)d_in[6];
  const float* WK_b = (const float*)d_in[7];
  const float* WV_w = (const float*)d_in[8];
  const float* WV_b = (const float*)d_in[9];
  const float* fc_w = (const float*)d_in[10];
  const float* fc_b = (const float*)d_in[11];

  float* out  = (float*)d_out;                       // [B,S,D] fp32
  float* attn = out + (long)B_ * S_ * D_;            // [B,H,S,S] fp32

  const long per = (long)B_ * H_ * S_ * DK_;         // 4M elements each
  __bf16* qs = (__bf16*)d_ws;                        // [BH,S,DK] bf16 (pre-scaled)
  __bf16* ks = qs + per;                             // [BH,S,DK] bf16
  __bf16* vT = ks + per;                             // [BH,DK,S] bf16
  __bf16* ao = vT + per;                             // [B*S, D]  bf16

  dim3 gProj(D_ / 64, (B_ * S_) / 32);
  proj_kernel<<<gProj, 32, 0, stream>>>(Q, WQ_w, WQ_b, qs, 0, 0.125f); // 1/sqrt(64)
  proj_kernel<<<gProj, 32, 0, stream>>>(K, WK_w, WK_b, ks, 1, 1.0f);
  proj_kernel<<<gProj, 32, 0, stream>>>(V, WV_w, WV_b, vT, 2, 1.0f);

  dim3 gS(S_ / 64, S_ / 32, B_ * H_);
  scores_kernel<<<gS, 32, 0, stream>>>(qs, ks, attn);

  const int rows = B_ * H_ * S_;
  softmax_kernel<<<rows / 4, 128, 0, stream>>>(attn, mask);

  dim3 gAV(1, S_ / 32, B_ * H_);
  attnv_kernel<<<gAV, 32, 0, stream>>>(attn, vT, ao);

  dim3 gFC(D_ / 64, (B_ * S_) / 32);
  fc_kernel<<<gFC, 32, 0, stream>>>(ao, fc_w, fc_b, out);
}
